// GQACausalSelfAttention_85091892068547
// MI455X (gfx1250) — matrix operand order repaired and checked by reading, not compiled, not run on hardware
//
#include <hip/hip_runtime.h>
#include <hip/hip_bf16.h>
#include <cstdint>

// Problem constants (match reference)
#define BATCH 4
#define TSEQ  1024
#define EDIM  2048
#define HEADS 32
#define GRP   8
#define DKH   64
#define KDIM  2048
#define MROWS 4096      // BATCH*TSEQ

typedef __attribute__((ext_vector_type(16))) __bf16 v16bf;
typedef __attribute__((ext_vector_type(8)))  __bf16 v8bf;
typedef __attribute__((ext_vector_type(8)))  float  v8f;

__device__ inline __bf16 f2bf(float f) {
  unsigned u = __builtin_bit_cast(unsigned, f);
  u += 0x7FFFu + ((u >> 16) & 1u);                 // round-to-nearest-even
  unsigned short hs = (unsigned short)(u >> 16);
  return __builtin_bit_cast(__bf16, hs);
}

__device__ inline v8f wmma_bf16(v16bf a, v16bf b, v8f c) {
  return __builtin_amdgcn_wmma_f32_16x16x32_bf16(
      false, a, false, b, (short)0, c, false, false);
}

// ---- CDNA5 async global->LDS copy (ASYNCcnt-tracked) ----------------------
__device__ inline void async_b128(const void* g, void* l) {
  uint32_t lo = (uint32_t)(uintptr_t)l;           // LDS byte offset (addr[31:0])
  asm volatile("global_load_async_to_lds_b128 %0, %1, off"
               :: "v"(lo), "v"(g) : "memory");
}
#define ASYNC_WAIT_4() asm volatile("s_wait_asynccnt 0x4" ::: "memory")
#define ASYNC_WAIT_0() asm volatile("s_wait_asynccnt 0x0" ::: "memory")

// A-frag (16x32 MxK) from LDS row (K contiguous). lane: m=l&15 (row ptr), h=l>>4
__device__ inline v16bf load_afrag(const __bf16* row, int h, int koff) {
  v16bf a;
#pragma unroll
  for (int j = 0; j < 8; ++j) {
    int kk = koff + 16 * (j >> 2) + 8 * h + 2 * (j & 3);
    a[2 * j]     = row[kk];
    a[2 * j + 1] = row[kk + 1];
  }
  return a;
}
// B-frag (32x16 KxN) from N-major LDS row (K contiguous). lane: n=l&15, h=l>>4
__device__ inline v16bf load_bfrag(const __bf16* row, int h, int koff) {
  v16bf b;
#pragma unroll
  for (int j = 0; j < 8; ++j) {
    int kk = koff + 16 * (j >> 2) + 8 * h + 2 * (j & 3);
    b[2 * j]     = row[kk];
    b[2 * j + 1] = row[kk + 1];
  }
  return b;
}

// ---------------------------------------------------------------------------
// One-time fp32 -> bf16 convert (contiguous)
// ---------------------------------------------------------------------------
__global__ __launch_bounds__(256) void cvt_bf16(const float* __restrict__ in,
                                                __bf16* __restrict__ out) {
  size_t i = ((size_t)blockIdx.x * 256 + threadIdx.x) * 8;
  float4 f0 = *(const float4*)(in + i);
  float4 f1 = *(const float4*)(in + i + 4);
  v8bf v;
  v[0] = f2bf(f0.x); v[1] = f2bf(f0.y); v[2] = f2bf(f0.z); v[3] = f2bf(f0.w);
  v[4] = f2bf(f1.x); v[5] = f2bf(f1.y); v[6] = f2bf(f1.z); v[7] = f2bf(f1.w);
  *(v8bf*)(out + i) = v;
}

// One-time fp32 [K][N] -> bf16 [N][K] convert+transpose (32x32 LDS tiles)
__global__ __launch_bounds__(256) void cvt_tr_bf16(const float* __restrict__ in,
                                                   __bf16* __restrict__ out,
                                                   int K, int N) {
  __shared__ __bf16 tile[32][33];
  int k0 = blockIdx.y * 32, n0 = blockIdx.x * 32;
  int r = threadIdx.x >> 3;            // 0..31
  int c = (threadIdx.x & 7) * 4;       // 0..28
  float4 f = *(const float4*)(in + (size_t)(k0 + r) * N + n0 + c);
  tile[r][c + 0] = f2bf(f.x); tile[r][c + 1] = f2bf(f.y);
  tile[r][c + 2] = f2bf(f.z); tile[r][c + 3] = f2bf(f.w);
  __syncthreads();
  __bf16* dst = out + (size_t)(n0 + r) * K + k0 + c;
  dst[0] = tile[c + 0][r]; dst[1] = tile[c + 1][r];
  dst[2] = tile[c + 2][r]; dst[3] = tile[c + 3][r];
}

// ---------------------------------------------------------------------------
// bf16 GEMM: out = A[M,K] @ Wt[N,K]^T + bias. Block 128x128, 8 waves,
// wave tile 32x64 (8 WMMA / K-step of 32). Async double-buffered staging.
// mode 0: bout0 = q bf16 [B,H,T,dk]
// mode 1: out0/out1 = k/v fp32 [B,G,T,dk]; bout0 = k bf16; bout1 = v^T bf16 [B,G,dk,T]
// mode 2: out0 = y fp32 [M,N]
// ---------------------------------------------------------------------------
#define GTM 128
#define GTN 128
#define GTK 32
#define GLD 40          // row stride (bf16): 80B, 16B-aligned, conflict-free

__global__ __launch_bounds__(256) void gqa_gemm_bf16(
    const __bf16* __restrict__ A, const __bf16* __restrict__ Wt,
    const float* __restrict__ bias, float* __restrict__ out0,
    float* __restrict__ out1, __bf16* __restrict__ bout0,
    __bf16* __restrict__ bout1, int mode, int N) {
  __shared__ __bf16 a_lds[2][GTM][GLD];
  __shared__ __bf16 b_lds[2][GTN][GLD];

  const int tid   = threadIdx.x;
  const int lane  = tid & 31;
  const int wave  = tid >> 5;
  const int waveM = wave & 3;            // 4 waves x 32 rows
  const int waveN = wave >> 2;           // 2 waves x 64 cols
  const int l16   = lane & 15;
  const int h     = lane >> 4;
  const int tileM = blockIdx.y * GTM;
  const int tileN = blockIdx.x * GTN;

  // staging assignment: row r = tid>>1 (0..127), col half c = (tid&1)*16
  const int sr = tid >> 1, sc = (tid & 1) * 16;
  const __bf16* ag = A  + (size_t)(tileM + sr) * KDIM + sc;
  const __bf16* bg = Wt + (size_t)(tileN + sr) * KDIM + sc;

  v8f acc[2][4] = {};

  // issue k-step 0 into buffer 0
  async_b128(ag,     &a_lds[0][sr][sc]);
  async_b128(ag + 8, &a_lds[0][sr][sc + 8]);
  async_b128(bg,     &b_lds[0][sr][sc]);
  async_b128(bg + 8, &b_lds[0][sr][sc + 8]);

  const int NIT = KDIM / GTK;
  for (int it = 0; it < NIT; ++it) {
    const int cur = it & 1;
    if (it + 1 < NIT) {
      const __bf16* ag2 = ag + (size_t)(it + 1) * GTK;
      const __bf16* bg2 = bg + (size_t)(it + 1) * GTK;
      async_b128(ag2,     &a_lds[cur ^ 1][sr][sc]);
      async_b128(ag2 + 8, &a_lds[cur ^ 1][sr][sc + 8]);
      async_b128(bg2,     &b_lds[cur ^ 1][sr][sc]);
      async_b128(bg2 + 8, &b_lds[cur ^ 1][sr][sc + 8]);
      ASYNC_WAIT_4();                    // group(it) complete (in-order)
    } else {
      ASYNC_WAIT_0();
    }
    __syncthreads();

    v16bf a0 = load_afrag(&a_lds[cur][waveM * 32 + l16][0], h, 0);
    v16bf a1 = load_afrag(&a_lds[cur][waveM * 32 + 16 + l16][0], h, 0);
#pragma unroll
    for (int t = 0; t < 4; ++t) {
      v16bf b = load_bfrag(&b_lds[cur][waveN * 64 + t * 16 + l16][0], h, 0);
      acc[0][t] = wmma_bf16(a0, b, acc[0][t]);
      acc[1][t] = wmma_bf16(a1, b, acc[1][t]);
    }
    __syncthreads();
  }

  // Epilogue. C/D layout: VGPR r -> row (r + 8*h), col l16 within each tile.
#pragma unroll
  for (int mi = 0; mi < 2; ++mi) {
#pragma unroll
    for (int t = 0; t < 4; ++t) {
      int n = tileN + waveN * 64 + t * 16 + l16;
      float bv = bias[n];
#pragma unroll
      for (int r = 0; r < 8; ++r) {
        int row = tileM + waveM * 32 + mi * 16 + 8 * h + r;
        float val = acc[mi][t][r] + bv;
        int b = row >> 10, tt = row & (TSEQ - 1);
        if (mode == 0) {                   // q bf16 [B,H,T,dk]
          int hh = n >> 6, d = n & 63;
          bout0[(((size_t)(b * HEADS + hh)) * TSEQ + tt) * DKH + d] = f2bf(val);
        } else if (mode == 1) {            // k/v fp32 outs + bf16 copies
          if (n < GRP * DKH) {
            int g = n >> 6, d = n & 63;
            size_t idx = (((size_t)(b * GRP + g)) * TSEQ + tt) * DKH + d;
            __builtin_nontemporal_store(val, &out0[idx]);
            bout0[idx] = f2bf(val);
          } else {
            int n2 = n - GRP * DKH;
            int g = n2 >> 6, d = n2 & 63;
            size_t idx = (((size_t)(b * GRP + g)) * TSEQ + tt) * DKH + d;
            __builtin_nontemporal_store(val, &out1[idx]);
            // v transposed bf16 [B,G,dk,T]
            bout1[(((size_t)(b * GRP + g)) * DKH + d) * TSEQ + tt] = f2bf(val);
          }
        } else {                           // y fp32 [M,N]
          __builtin_nontemporal_store(val, &out0[(size_t)row * N + n]);
        }
      }
    }
  }
}

// ---------------------------------------------------------------------------
// Flash causal attention, all-bf16 operands, async double-buffered K/V tiles.
// Block = 64 q rows of one (b,h); 4 waves x 16 rows; kv chunks of 32.
// ---------------------------------------------------------------------------
__global__ __launch_bounds__(128) void gqa_flash_attn(
    const __bf16* __restrict__ qb, const __bf16* __restrict__ kb,
    const __bf16* __restrict__ vbt, __bf16* __restrict__ ab) {
  __shared__ __bf16 q_lds[64][72];        // [qrow][d]
  __shared__ __bf16 k_lds[2][32][72];     // [kv][d]
  __shared__ __bf16 v_lds[2][64][40];     // [d][kv]  (from pre-transposed V)
  __shared__ __bf16 p_lds[4][16][34];     // per-wave P relayout C->A

  const int tid  = threadIdx.x;
  const int lane = tid & 31;
  const int wave = tid >> 5;
  const int l16  = lane & 15;
  const int h    = lane >> 4;
  const int b     = blockIdx.z;
  const int hh    = blockIdx.y;
  const int qbase = blockIdx.x * 64;
  const int g     = hh & (GRP - 1);       // head h uses group h % G

  const __bf16* qptr = qb  + (((size_t)(b * HEADS + hh)) * TSEQ + qbase) * DKH;
  const __bf16* kptr = kb  + ((size_t)(b * GRP + g)) * TSEQ * DKH;
  const __bf16* vptr = vbt + ((size_t)(b * GRP + g)) * DKH * TSEQ;

  // staging assignments
  const int qr = tid >> 1, qc = (tid & 1) * 32;          // Q: 4 x b128
  const int ki = tid >> 2, kc = (tid & 2 ? 32 : 0) + (tid & 1) * 16; // K: 2 x b128
  const int vd = tid >> 1, vc = (tid & 1) * 16;          // V^T: 2 x b128

  // issue Q + KV chunk 0
  {
    const __bf16* qs = qptr + (size_t)qr * DKH + qc;
#pragma unroll
    for (int q = 0; q < 4; ++q) async_b128(qs + q * 8, &q_lds[qr][qc + q * 8]);
    const __bf16* ks = kptr + (size_t)ki * DKH + kc;
    async_b128(ks,     &k_lds[0][ki][kc]);
    async_b128(ks + 8, &k_lds[0][ki][kc + 8]);
    const __bf16* vs = vptr + (size_t)vd * TSEQ + vc;
    async_b128(vs,     &v_lds[0][vd][vc]);
    async_b128(vs + 8, &v_lds[0][vd][vc + 8]);
  }
  ASYNC_WAIT_4();                         // Q done (chunk0 may be in flight)
  __syncthreads();
  const v16bf qa0 = load_afrag(&q_lds[wave * 16 + l16][0], h, 0);
  const v16bf qa1 = load_afrag(&q_lds[wave * 16 + l16][0], h, 32);

  float m_i[8], l_i[8];
#pragma unroll
  for (int r = 0; r < 8; ++r) { m_i[r] = -3.0e38f; l_i[r] = 0.0f; }
  v8f accO[4] = {};

  const int NCH = qbase / 32 + 2;         // causal: chunks covering [0, qbase+64)
  for (int it = 0; it < NCH; ++it) {
    const int cur = it & 1;
    if (it + 1 < NCH) {
      const int kb2 = (it + 1) * 32;
      const __bf16* ks = kptr + (size_t)(kb2 + ki) * DKH + kc;
      async_b128(ks,     &k_lds[cur ^ 1][ki][kc]);
      async_b128(ks + 8, &k_lds[cur ^ 1][ki][kc + 8]);
      const __bf16* vs = vptr + (size_t)vd * TSEQ + kb2 + vc;
      async_b128(vs,     &v_lds[cur ^ 1][vd][vc]);
      async_b128(vs + 8, &v_lds[cur ^ 1][vd][vc + 8]);
      ASYNC_WAIT_4();
    } else {
      ASYNC_WAIT_0();
    }
    __syncthreads();

    // S = Q @ K^T (two 16-col kv tiles)
    v8f s0 = {}, s1 = {};
    {
      v16bf b0 = load_bfrag(&k_lds[cur][l16][0], h, 0);
      v16bf b1 = load_bfrag(&k_lds[cur][l16][0], h, 32);
      s0 = wmma_bf16(qa0, b0, s0);
      s0 = wmma_bf16(qa1, b1, s0);
      v16bf b2 = load_bfrag(&k_lds[cur][16 + l16][0], h, 0);
      v16bf b3 = load_bfrag(&k_lds[cur][16 + l16][0], h, 32);
      s1 = wmma_bf16(qa0, b2, s1);
      s1 = wmma_bf16(qa1, b3, s1);
    }

    // online softmax (row cols live in one 16-lane half)
    const int kbase = it * 32;
    const int qrow0 = qbase + wave * 16 + 8 * h;
    const int kv0 = kbase + l16, kv1 = kbase + 16 + l16;
#pragma unroll
    for (int r = 0; r < 8; ++r) {
      int qp = qrow0 + r;
      float x0 = s0[r] * 0.125f;          // 1/sqrt(64)
      float x1 = s1[r] * 0.125f;
      if (kv0 > qp) x0 = -3.0e38f;
      if (kv1 > qp) x1 = -3.0e38f;
      float cm = fmaxf(x0, x1);
      cm = fmaxf(cm, __shfl_xor(cm, 1));
      cm = fmaxf(cm, __shfl_xor(cm, 2));
      cm = fmaxf(cm, __shfl_xor(cm, 4));
      cm = fmaxf(cm, __shfl_xor(cm, 8));
      float mn = fmaxf(m_i[r], cm);
      float alpha = (m_i[r] < -1.0e37f) ? 0.0f : __expf(m_i[r] - mn);
      m_i[r] = mn;
      float p0 = __expf(x0 - mn);
      float p1 = __expf(x1 - mn);
      float rs = p0 + p1;
      rs += __shfl_xor(rs, 1);
      rs += __shfl_xor(rs, 2);
      rs += __shfl_xor(rs, 4);
      rs += __shfl_xor(rs, 8);
      l_i[r] = l_i[r] * alpha + rs;
#pragma unroll
      for (int t = 0; t < 4; ++t) accO[t][r] *= alpha;
      p_lds[wave][8 * h + r][l16]      = f2bf(p0);
      p_lds[wave][8 * h + r][16 + l16] = f2bf(p1);
    }

    // O += P @ V
    v16bf pa = load_afrag(&p_lds[wave][l16][0], h, 0);
#pragma unroll
    for (int t = 0; t < 4; ++t) {
      v16bf vb = load_bfrag(&v_lds[cur][t * 16 + l16][0], h, 0);
      accO[t] = wmma_bf16(pa, vb, accO[t]);
    }
    __syncthreads();
  }

  // normalize + store merged-head bf16 [B,T,H*dk]
#pragma unroll
  for (int r = 0; r < 8; ++r) {
    int tq = qbase + wave * 16 + 8 * h + r;
    float inv = 1.0f / l_i[r];
#pragma unroll
    for (int t = 0; t < 4; ++t) {
      int d = t * 16 + l16;
      ab[((size_t)(b * TSEQ + tq)) * EDIM + hh * DKH + d] = f2bf(accO[t][r] * inv);
    }
  }
}

extern "C" void kernel_launch(void* const* d_in, const int* in_sizes, int n_in,
                              void* d_out, int out_size, void* d_ws, size_t ws_size,
                              hipStream_t stream) {
  const float* x   = (const float*)d_in[0];
  const float* Wq  = (const float*)d_in[1];
  const float* bq  = (const float*)d_in[2];
  const float* Wkv = (const float*)d_in[3];
  const float* bkv = (const float*)d_in[4];
  const float* Wc  = (const float*)d_in[5];
  const float* bc  = (const float*)d_in[6];

  float* y_out = (float*)d_out;                                   // [B,T,E]
  float* k_out = y_out + (size_t)MROWS * EDIM;                    // [B,G,T,dk]
  float* v_out = k_out + (size_t)BATCH * GRP * TSEQ * DKH;        // [B,G,T,dk]

  // bf16 workspace layout
  __bf16* xb   = (__bf16*)d_ws;                                   // [M,K]
  __bf16* wqb  = xb   + (size_t)MROWS * KDIM;                     // [2048][2048]
  __bf16* wkvb = wqb  + (size_t)EDIM * KDIM;                      // [1024][2048]
  __bf16* wcb  = wkvb + (size_t)(2 * GRP * DKH) * KDIM;           // [2048][2048]
  __bf16* qbw  = wcb  + (size_t)EDIM * KDIM;                      // [B,H,T,dk]
  __bf16* kbw  = qbw  + (size_t)MROWS * EDIM;                     // [B,G,T,dk]
  __bf16* vbtw = kbw  + (size_t)BATCH * GRP * TSEQ * DKH;         // [B,G,dk,T]
  __bf16* abw  = vbtw + (size_t)BATCH * GRP * TSEQ * DKH;         // [B,T,E]

  // 0) one-time conversions / weight transposes
  cvt_bf16<<<dim3((MROWS * (size_t)KDIM) / (256 * 8)), 256, 0, stream>>>(x, xb);
  cvt_tr_bf16<<<dim3(EDIM / 32, KDIM / 32), 256, 0, stream>>>(Wq, wqb, KDIM, EDIM);
  cvt_tr_bf16<<<dim3((2 * GRP * DKH) / 32, KDIM / 32), 256, 0, stream>>>(
      Wkv, wkvb, KDIM, 2 * GRP * DKH);
  cvt_tr_bf16<<<dim3(EDIM / 32, KDIM / 32), 256, 0, stream>>>(Wc, wcb, KDIM, EDIM);

  // 1) Q projection (bf16 out)
  gqa_gemm_bf16<<<dim3(EDIM / GTN, MROWS / GTM), 256, 0, stream>>>(
      xb, wqb, bq, nullptr, nullptr, qbw, nullptr, 0, EDIM);
  // 2) KV projection (fp32 k/v outputs + bf16 k + bf16 v^T)
  gqa_gemm_bf16<<<dim3((2 * GRP * DKH) / GTN, MROWS / GTM), 256, 0, stream>>>(
      xb, wkvb, bkv, k_out, v_out, kbw, vbtw, 1, 2 * GRP * DKH);
  // 3) flash causal attention
  gqa_flash_attn<<<dim3(TSEQ / 64, HEADS, BATCH), dim3(128), 0, stream>>>(
      qbw, kbw, vbtw, abw);
  // 4) output projection (fp32 y)
  gqa_gemm_bf16<<<dim3(EDIM / GTN, MROWS / GTM), 256, 0, stream>>>(
      abw, wcb, bc, y_out, nullptr, nullptr, nullptr, 2, EDIM);
}
